// EncoderSimilarityt2i_78280073937197
// MI455X (gfx1250) — matrix operand order repaired and checked by reading, not compiled, not run on hardware
//
#include <hip/hip_runtime.h>

#define DD    1024
#define NB    64      // n_image
#define NS    36      // regions
#define NC    64      // n_caption
#define LMAXW 32
#define SMOOTHF 9.0f
#define LLSEF   6.0f
#define EPSF    1e-8f

typedef __attribute__((ext_vector_type(16))) __bf16 v16bf;
typedef __attribute__((ext_vector_type(8)))  float  v8f;

union FragU { uint4 q[2]; v16bf v; };
union AccU  { v8f v; float f[8]; };

__device__ __forceinline__ unsigned f2bf(float f) {
  unsigned u = __builtin_bit_cast(unsigned, f);
  return (u + 0x7FFFu + ((u >> 16) & 1u)) >> 16;   // RNE fp32 -> bf16
}
__device__ __forceinline__ unsigned pk2(float lo, float hi) {
  return f2bf(lo) | (f2bf(hi) << 16);
}
__device__ __forceinline__ uint4 cvt8(const float* p) {
  float4 a = ((const float4*)p)[0];
  float4 b = ((const float4*)p)[1];
  return make_uint4(pk2(a.x, a.y), pk2(a.z, a.w), pk2(b.x, b.y), pk2(b.z, b.w));
}

// ---- WMMA fragment loaders -------------------------------------------------
// A-matrix (16xK, bf16): lane = 16*h + m; per lane two contiguous 16B chunks:
//   halves 0..7  -> K = k0 + 8h .. +7
//   halves 8..15 -> K = k0 + 16 + 8h .. +7
__device__ __forceinline__ v16bf a_frag_lds(const unsigned short* s, int row0, int stride, int k0) {
  int lane = threadIdx.x & 31;
  int m = lane & 15, h = lane >> 4;
  const unsigned short* p = s + (row0 + m) * stride + k0 + 8 * h;
  FragU f;
  f.q[0] = *(const uint4*)p;
  f.q[1] = *(const uint4*)(p + 16);
  return f.v;
}
// B-matrix (Kx16, bf16): lane = 16*h + n; K = k0 + 8h .. +7 and k0 + 16 + 8h .. +7 of column n.
// Column n of B == row n of the transposed operand -> contiguous loads.
__device__ __forceinline__ v16bf b_frag_lds(const unsigned short* s, int row0, int stride, int k0) {
  int lane = threadIdx.x & 31;
  int n = lane & 15, h = lane >> 4;
  const unsigned short* p = s + (row0 + n) * stride + k0 + 8 * h;
  FragU f;
  f.q[0] = *(const uint4*)p;
  f.q[1] = *(const uint4*)(p + 16);
  return f.v;
}
__device__ __forceinline__ v16bf a_frag_g(const float* base, int row0, int nrows, int ld, int k0) {
  int lane = threadIdx.x & 31;
  int m = lane & 15, h = lane >> 4;
  FragU f;
  int r = row0 + m;
  if (r < nrows) {
    const float* p = base + (size_t)r * ld + k0 + 8 * h;
    f.q[0] = cvt8(p);
    f.q[1] = cvt8(p + 16);
  } else {
    f.q[0] = make_uint4(0,0,0,0); f.q[1] = make_uint4(0,0,0,0);
  }
  return f.v;
}
__device__ __forceinline__ v16bf b_frag_g(const float* base, int row0, int nrows, int ld, int k0) {
  int lane = threadIdx.x & 31;
  int n = lane & 15, h = lane >> 4;
  FragU f;
  int r = row0 + n;
  if (r < nrows) {
    const float* p = base + (size_t)r * ld + k0 + 8 * h;
    f.q[0] = cvt8(p);
    f.q[1] = cvt8(p + 16);
  } else {
    f.q[0] = make_uint4(0,0,0,0); f.q[1] = make_uint4(0,0,0,0);
  }
  return f.v;
}

#define WMMA_BF16(a, b, c) \
  __builtin_amdgcn_wmma_f32_16x16x32_bf16(false, (a), false, (b), (short)0, (c), false, false)

// ---- Kernel 1: energy = captions_flat(2048x1024) @ W_g^T ------------------
__global__ __launch_bounds__(256) void k_energy(const float* __restrict__ cap,
                                                const float* __restrict__ wg,
                                                unsigned short* __restrict__ energy) {
  __shared__ unsigned short sA[64 * 136];
  __shared__ unsigned short sB[32 * 136];
  const int tid = threadIdx.x;
  const int wave = tid >> 5;
  const int m0 = blockIdx.x * 64;
  const int n0 = blockIdx.y * 32;
  const int mt = wave >> 1, nt = wave & 1;
  v8f acc = {0.f,0.f,0.f,0.f,0.f,0.f,0.f,0.f};
  for (int ch = 0; ch < 8; ++ch) {
    const int k0 = ch * 128;
    __syncthreads();
    for (int idx = tid; idx < 64 * 32; idx += 256) {
      int row = idx >> 5, u = idx & 31;
      float4 v = *(const float4*)(cap + (size_t)(m0 + row) * DD + k0 + u * 4);
      *(uint2*)&sA[row * 136 + u * 4] = make_uint2(pk2(v.x, v.y), pk2(v.z, v.w));
    }
    for (int idx = tid; idx < 32 * 32; idx += 256) {
      int row = idx >> 5, u = idx & 31;
      float4 v = *(const float4*)(wg + (size_t)(n0 + row) * DD + k0 + u * 4);
      *(uint2*)&sB[row * 136 + u * 4] = make_uint2(pk2(v.x, v.y), pk2(v.z, v.w));
    }
    __syncthreads();
    for (int ks = 0; ks < 4; ++ks) {
      v16bf a = a_frag_lds(sA, mt * 16, 136, ks * 32);
      v16bf b = b_frag_lds(sB, nt * 16, 136, ks * 32);
      acc = WMMA_BF16(a, b, acc);
    }
  }
  AccU A; A.v = acc;
  int lane = tid & 31; int h = lane >> 4; int nn = lane & 15;
  #pragma unroll
  for (int r = 0; r < 8; ++r) {
    int m = m0 + mt * 16 + r + 8 * h;
    int n = n0 + nt * 16 + nn;
    energy[(size_t)m * DD + n] = (unsigned short)f2bf(A.f[r]);
  }
}

// ---- Kernel 2: per-image gram (36x36) = img_b @ img_b^T --------------------
__global__ __launch_bounds__(288) void k_gram(const float* __restrict__ images,
                                              float* __restrict__ gram) {
  const int b = blockIdx.x;
  const int wave = threadIdx.x >> 5;      // 0..8 -> 3x3 tiles of 16
  const int mt = wave / 3, nt = wave % 3;
  const float* im = images + (size_t)b * NS * DD;
  v8f acc = {0.f,0.f,0.f,0.f,0.f,0.f,0.f,0.f};
  for (int ks = 0; ks < 32; ++ks) {
    v16bf a  = a_frag_g(im, mt * 16, NS, DD, ks * 32);
    v16bf bb = b_frag_g(im, nt * 16, NS, DD, ks * 32);
    acc = WMMA_BF16(a, bb, acc);
  }
  AccU A; A.v = acc;
  int lane = threadIdx.x & 31; int h = lane >> 4; int nn = lane & 15;
  #pragma unroll
  for (int r = 0; r < 8; ++r) {
    int M = mt * 16 + r + 8 * h;
    int N = nt * 16 + nn;
    if (M < NS && N < NS) gram[(size_t)b * NS * NS + M * NS + N] = A.f[r];
  }
}

// ---- Kernel 3: caption word norms -----------------------------------------
__global__ __launch_bounds__(256) void k_w1(const float* __restrict__ cap,
                                            float* __restrict__ w1) {
  int r = blockIdx.x * 256 + threadIdx.x;
  if (r < NC * LMAXW) {
    const float* p = cap + (size_t)r * DD;
    float s = 0.f;
    for (int d = 0; d < DD; ++d) { float v = p[d]; s += v * v; }
    w1[r] = sqrtf(s);
  }
}

// ---- Kernel 4: per (image b, caption i) similarity -------------------------
__global__ __launch_bounds__(256) void k_main(
    const float* __restrict__ images, const float* __restrict__ captions,
    const int* __restrict__ cap_lens, const unsigned short* __restrict__ energy,
    const float* __restrict__ gram, const float* __restrict__ w1,
    float* __restrict__ out) {
  __shared__ unsigned short s_img[48 * 136];   // rows >= 36 zero-padded
  __shared__ unsigned short s_eng[32 * 136];
  __shared__ unsigned short s_cap[32 * 136];
  __shared__ float s_attn[48 * 33];
  __shared__ float s_G[48 * 33];
  __shared__ float s_gram[NS * NS];
  __shared__ float s_z[LMAXW];
  const int b = blockIdx.x, i = blockIdx.y;
  const int tid = threadIdx.x;
  const int wave = tid >> 5;
  const float* imgb = images + (size_t)b * NS * DD;
  const float* capi = captions + (size_t)i * LMAXW * DD;
  const unsigned short* engi = energy + (size_t)i * LMAXW * DD;

  // 12 tile-tasks on 8 waves: 6 attn tiles (img @ energy^T), 6 G tiles (img @ cap^T)
  int tgt0, mt0, nt0;
  if (wave < 6) { tgt0 = 0; mt0 = wave >> 1; nt0 = wave & 1; }
  else          { tgt0 = 1; mt0 = 0;         nt0 = wave - 6; }
  const bool has2 = (wave < 4);
  const int mt1 = (wave + 2) >> 1, nt1 = (wave + 2) & 1;   // always tgt=1

  v8f acc0 = {0.f,0.f,0.f,0.f,0.f,0.f,0.f,0.f};
  v8f acc1 = {0.f,0.f,0.f,0.f,0.f,0.f,0.f,0.f};
  for (int ch = 0; ch < 8; ++ch) {
    const int k0 = ch * 128;
    __syncthreads();
    for (int idx = tid; idx < 48 * 32; idx += 256) {
      int row = idx >> 5, u = idx & 31;
      uint2 pkv = make_uint2(0u, 0u);
      if (row < NS) {
        float4 v = *(const float4*)(imgb + (size_t)row * DD + k0 + u * 4);
        pkv = make_uint2(pk2(v.x, v.y), pk2(v.z, v.w));
      }
      *(uint2*)&s_img[row * 136 + u * 4] = pkv;
    }
    for (int idx = tid; idx < 32 * 16; idx += 256) {
      int row = idx >> 4, u = idx & 15;
      uint4 q = *(const uint4*)(engi + (size_t)row * DD + k0 + u * 8);
      *(uint4*)&s_eng[row * 136 + u * 8] = q;
    }
    for (int idx = tid; idx < 32 * 32; idx += 256) {
      int row = idx >> 5, u = idx & 31;
      float4 v = *(const float4*)(capi + (size_t)row * DD + k0 + u * 4);
      *(uint2*)&s_cap[row * 136 + u * 4] = make_uint2(pk2(v.x, v.y), pk2(v.z, v.w));
    }
    if (ch == 0) {
      for (int idx = tid; idx < NS * NS; idx += 256)
        s_gram[idx] = gram[(size_t)b * NS * NS + idx];
    }
    __syncthreads();
    for (int ks = 0; ks < 4; ++ks) {
      const int kk = ks * 32;
      v16bf a0 = a_frag_lds(s_img, mt0 * 16, 136, kk);
      v16bf b0 = b_frag_lds(tgt0 ? s_cap : s_eng, nt0 * 16, 136, kk);
      acc0 = WMMA_BF16(a0, b0, acc0);
      if (has2) {
        v16bf a1 = a_frag_lds(s_img, mt1 * 16, 136, kk);
        v16bf b1 = b_frag_lds(s_cap, nt1 * 16, 136, kk);
        acc1 = WMMA_BF16(a1, b1, acc1);
      }
    }
  }
  __syncthreads();
  {
    AccU A; A.v = acc0;
    const int lane = tid & 31; const int h = lane >> 4; const int nn = lane & 15;
    float* dst = tgt0 ? s_G : s_attn;
    #pragma unroll
    for (int r = 0; r < 8; ++r)
      dst[(mt0 * 16 + r + 8 * h) * 33 + nt0 * 16 + nn] = A.f[r];
    if (has2) {
      AccU B2; B2.v = acc1;
      #pragma unroll
      for (int r = 0; r < 8; ++r)
        s_G[(mt1 * 16 + r + 8 * h) * 33 + nt1 * 16 + nn] = B2.f[r];
    }
  }
  __syncthreads();
  const int len = cap_lens[i];
  // leaky-relu + word mask + l2norm over words, per region (rows)
  if (tid < NS) {
    float vals[LMAXW];
    float ss = 0.f;
    #pragma unroll
    for (int l = 0; l < LMAXW; ++l) {
      float a = s_attn[tid * 33 + l];
      a = (a > 0.f) ? a : 0.1f * a;
      a = (l < len) ? a : 0.f;
      vals[l] = a;
      ss += a * a;
    }
    float inv = 1.f / (sqrtf(ss) + EPSF);
    #pragma unroll
    for (int l = 0; l < LMAXW; ++l) s_attn[tid * 33 + l] = vals[l] * inv;
  }
  __syncthreads();
  // softmax over regions per word + cosine sim via gram quadratic form
  if (tid < LMAXW) {
    float m = -1e30f;
    for (int s = 0; s < NS; ++s) m = fmaxf(m, s_attn[s * 33 + tid]);
    float den = 0.f;
    for (int s = 0; s < NS; ++s) {
      float e = expf(SMOOTHF * (s_attn[s * 33 + tid] - m));
      s_attn[s * 33 + tid] = e;
      den += e;
    }
    float invd = 1.f / den;
    float w12 = 0.f;
    for (int s = 0; s < NS; ++s) {
      float w = s_attn[s * 33 + tid] * invd;
      s_attn[s * 33 + tid] = w;
      w12 += w * s_G[s * 33 + tid];           // cap[l] . wctx[l]
    }
    float wn2 = 0.f;                           // ||wctx[l]||^2 = a^T Gram a
    for (int sp = 0; sp < NS; ++sp) {
      float t = 0.f;
      for (int s = 0; s < NS; ++s) t += s_attn[s * 33 + tid] * s_gram[s * NS + sp];
      wn2 += t * s_attn[sp * 33 + tid];
    }
    float w1v = w1[i * LMAXW + tid];
    float w2v = sqrtf(fmaxf(wn2, 0.f));
    float row_sim = w12 / fmaxf(w1v * w2v, EPSF);
    s_z[tid] = (tid < len) ? LLSEF * row_sim : -1e30f;
  }
  __syncthreads();
  if (tid == 0) {
    float m = -1e30f;
    for (int l = 0; l < LMAXW; ++l) m = fmaxf(m, s_z[l]);
    float ssum = 0.f;
    for (int l = 0; l < LMAXW; ++l) ssum += expf(s_z[l] - m);
    out[b * NC + i] = (m + logf(ssum)) / LLSEF;   // sims.T -> (image, caption)
  }
}

extern "C" void kernel_launch(void* const* d_in, const int* in_sizes, int n_in,
                              void* d_out, int out_size, void* d_ws, size_t ws_size,
                              hipStream_t stream) {
  (void)in_sizes; (void)n_in; (void)out_size; (void)ws_size;
  const float* images   = (const float*)d_in[0];   // (64, 36, 1024)
  const float* captions = (const float*)d_in[1];   // (64, 32, 1024)
  const int*   cap_lens = (const int*)d_in[2];     // (64,)
  const float* W_g      = (const float*)d_in[3];   // (1024, 1024)
  float* out = (float*)d_out;                      // (64, 64)

  char* ws = (char*)d_ws;
  unsigned short* energy = (unsigned short*)ws;                       // 2048*1024 bf16 = 4 MB
  float* gram = (float*)(ws + (size_t)2048 * 1024 * 2);               // 64*36*36 f32
  float* w1n  = (float*)(ws + (size_t)2048 * 1024 * 2 + (size_t)NB * NS * NS * 4);

  k_energy<<<dim3(32, 32), 256, 0, stream>>>(captions, W_g, energy);
  k_gram  <<<NB, 288, 0, stream>>>(images, gram);
  k_w1    <<<8, 256, 0, stream>>>(captions, w1n);
  k_main  <<<dim3(NB, NC), 256, 0, stream>>>(images, captions, cap_lens,
                                             energy, gram, w1n, out);
}
